// MemoryEfficientLoRA_33603824124646
// MI455X (gfx1250) — matrix operand order repaired and checked by reading, not compiled, not run on hardware
//
#include <hip/hip_runtime.h>
#include <hip/hip_bf16.h>

// ---------------------------------------------------------------------------
// Fused LoRA forward for MI455X (gfx1250, wave32, WMMA).
//   out = ((x @ A_dq^T) @ B_dq^T) * SCALING
// Single main kernel: each wave owns 16 rows of x; stage 1 accumulates
// h(16x64) in WMMA accumulators over K=4096; h is transposed C/D->A-fragment
// layout through 2 KB of LDS (cross-lane reshuffle) and consumed immediately
// by stage 2 over all 4096 output columns. h never touches HBM.
// HBM traffic = read x (256 MB) + write out (256 MB) -> ~22 us floor @23.3TB/s.
// Weights (1 MB bf16) are L2-resident. Dequant scales factor out of the
// reductions and are applied post-accumulation in f32.
// ---------------------------------------------------------------------------

typedef __attribute__((ext_vector_type(16))) __bf16 v16bf;
typedef __attribute__((ext_vector_type(8)))  __bf16 v8bf;
typedef __attribute__((ext_vector_type(8)))  float  v8f;
typedef __attribute__((ext_vector_type(4)))  float  v4f;

#define LORA_RANK 64
#define LORA_DIN  4096
#define LORA_DOUT 4096
#define LORA_M    16384          // B*S = 4*4096
#define LORA_SCALING 0.25f       // ALPHA/RANK = 16/64

// ---------------------------------------------------------------------------
// One-time f32 -> bf16 conversion of the small weight matrices (512 KB each).
// ---------------------------------------------------------------------------
__global__ __launch_bounds__(256)
void cvt_f32_bf16_kernel(const float* __restrict__ src,
                         __bf16* __restrict__ dst, int n) {
    int i = blockIdx.x * 256 + threadIdx.x;
    int stride = gridDim.x * 256;
    for (; i < n; i += stride) dst[i] = (__bf16)src[i];
}

// ---------------------------------------------------------------------------
// Fused kernel: grid = M/128 blocks, 256 threads (8 waves), wave w -> rows
// m0..m0+15. LDS: 8 waves * 16x64 bf16 h-tiles = 16 KB.
// ---------------------------------------------------------------------------
__global__ __launch_bounds__(256)
void lora_fused_kernel(const float*  __restrict__ x,       // [M, DIN] f32
                       const __bf16* __restrict__ Abf,     // [RANK, DIN] bf16
                       const __bf16* __restrict__ Bbf,     // [DOUT, RANK] bf16
                       const float*  __restrict__ a_scale, // [RANK]
                       const float*  __restrict__ b_scale, // [DOUT]
                       float*        __restrict__ out) {   // [M, DOUT] f32
    __shared__ __bf16 hlds[8][16][LORA_RANK];              // 16 KB

    const int lane = threadIdx.x & 31;
    const int wave = threadIdx.x >> 5;
    const int half = lane >> 4;          // 0: lanes 0-15, 1: lanes 16-31
    const int l16  = lane & 15;
    const int m0   = blockIdx.x * 128 + wave * 16;

    const float* xrow = x + (size_t)(m0 + l16) * LORA_DIN;

    // ======================= Stage 1: h = x @ A_dq^T =======================
    v8f acc[4];
    #pragma unroll
    for (int t = 0; t < 4; ++t) acc[t] = v8f{};

    // A-fragment raw x (f32) loads; 16x32 bf16 A layout: lane = row l16,
    // half 0 -> K kt+0..15, half 1 -> K kt+16..31 (matches B fragment).
    auto load_x = [&](int kt, v4f& x0, v4f& x1, v4f& x2, v4f& x3) {
        const float* p = xrow + kt + half * 16;
        x0 = *(const v4f*)(p);
        x1 = *(const v4f*)(p + 4);
        x2 = *(const v4f*)(p + 8);
        x3 = *(const v4f*)(p + 12);
    };
    // B-fragment (32x16 bf16): lane = col N, half 0 -> K kt..kt+15,
    // half 1 -> K kt+16..kt+31; 32 B contiguous per lane.
    auto load_w = [&](int kt, v16bf& b0, v16bf& b1, v16bf& b2, v16bf& b3) {
        const __bf16* p = Abf + (size_t)l16 * LORA_DIN + (kt + half * 16);
        b0 = *(const v16bf*)(p);
        b1 = *(const v16bf*)(p + 16 * LORA_DIN);
        b2 = *(const v16bf*)(p + 32 * LORA_DIN);
        b3 = *(const v16bf*)(p + 48 * LORA_DIN);
    };
    auto compute1 = [&](v4f x0, v4f x1, v4f x2, v4f x3,
                        v16bf b0, v16bf b1, v16bf b2, v16bf b3) {
        v16bf a;
        #pragma unroll
        for (int j = 0; j < 4; ++j) {
            a[j]      = (__bf16)x0[j];
            a[4 + j]  = (__bf16)x1[j];
            a[8 + j]  = (__bf16)x2[j];
            a[12 + j] = (__bf16)x3[j];
        }
        acc[0] = __builtin_amdgcn_wmma_f32_16x16x32_bf16(false, a, false, b0, (short)0, acc[0], false, false);
        acc[1] = __builtin_amdgcn_wmma_f32_16x16x32_bf16(false, a, false, b1, (short)0, acc[1], false, false);
        acc[2] = __builtin_amdgcn_wmma_f32_16x16x32_bf16(false, a, false, b2, (short)0, acc[2], false, false);
        acc[3] = __builtin_amdgcn_wmma_f32_16x16x32_bf16(false, a, false, b3, (short)0, acc[3], false, false);
    };

    // Software-pipelined K loop: loads for slab kt+32 issue before slab kt's
    // WMMAs, so the s_wait before each WMMA covers a full iteration of latency.
    v4f   cx0, cx1, cx2, cx3;
    v16bf cb0, cb1, cb2, cb3;
    load_x(0, cx0, cx1, cx2, cx3);
    load_w(0, cb0, cb1, cb2, cb3);
    for (int kt = 0; kt < LORA_DIN - 32; kt += 32) {
        v4f   nx0, nx1, nx2, nx3;
        v16bf nb0, nb1, nb2, nb3;
        load_x(kt + 32, nx0, nx1, nx2, nx3);
        load_w(kt + 32, nb0, nb1, nb2, nb3);
        __builtin_prefetch(xrow + kt + 64 + half * 8, 0, 1);  // global_prefetch_b8
        compute1(cx0, cx1, cx2, cx3, cb0, cb1, cb2, cb3);
        cx0 = nx0; cx1 = nx1; cx2 = nx2; cx3 = nx3;
        cb0 = nb0; cb1 = nb1; cb2 = nb2; cb3 = nb3;
    }
    compute1(cx0, cx1, cx2, cx3, cb0, cb1, cb2, cb3);

    // Epilogue: apply a_scale in f32, convert, park h tile in LDS.
    // C/D layout: VGPR j, lane -> row j + 8*half, col l16 (within 16-col tile).
    #pragma unroll
    for (int t = 0; t < 4; ++t) {
        const int r = t * 16 + l16;
        const float s = a_scale[r];
        #pragma unroll
        for (int j = 0; j < 8; ++j)
            hlds[wave][j + 8 * half][r] = (__bf16)(acc[t][j] * s);
    }
    __syncthreads();   // also fences the compiler across the LDS transpose

    // =================== Stage 2: out = h @ B_dq^T * s =====================
    // A-fragments for rank-64 (two 16x32 k-steps) read back transposed:
    // lane = row l16, contiguous r-runs of 8 bf16 (ds_load_b128 each pair).
    const int kb0 = half * 16;
    v8bf t0 = *(const v8bf*)&hlds[wave][l16][kb0];
    v8bf t1 = *(const v8bf*)&hlds[wave][l16][kb0 + 8];
    v8bf t2 = *(const v8bf*)&hlds[wave][l16][kb0 + 32];
    v8bf t3 = *(const v8bf*)&hlds[wave][l16][kb0 + 40];
    v16bf a0, a1;
    #pragma unroll
    for (int j = 0; j < 8; ++j) {
        a0[j] = t0[j]; a0[8 + j] = t1[j];
        a1[j] = t2[j]; a1[8 + j] = t3[j];
    }

    auto load_b = [&](int nt, v16bf& b0, v16bf& b1) {
        const __bf16* p = Bbf + (size_t)(nt * 16 + l16) * LORA_RANK + half * 16;
        b0 = *(const v16bf*)(p);        // r 0..31 slab
        b1 = *(const v16bf*)(p + 32);   // r 32..63 slab
    };
    auto emit = [&](int nt, v16bf b0, v16bf b1) {
        v8f o = v8f{};
        o = __builtin_amdgcn_wmma_f32_16x16x32_bf16(false, a0, false, b0, (short)0, o, false, false);
        o = __builtin_amdgcn_wmma_f32_16x16x32_bf16(false, a1, false, b1, (short)0, o, false, false);
        const int ocol = nt * 16 + l16;
        const float s = b_scale[ocol] * LORA_SCALING;
        #pragma unroll
        for (int j = 0; j < 8; ++j)
            out[(size_t)(m0 + j + 8 * half) * LORA_DOUT + ocol] = o[j] * s;
    };

    // Double-buffered N loop over all 4096 output columns (256 tiles of 16).
    v16bf pb0, pb1;
    load_b(0, pb0, pb1);
    for (int nt = 0; nt < LORA_DOUT / 16 - 1; ++nt) {
        v16bf nb0, nb1;
        load_b(nt + 1, nb0, nb1);
        emit(nt, pb0, pb1);
        pb0 = nb0; pb1 = nb1;
    }
    emit(LORA_DOUT / 16 - 1, pb0, pb1);
}

// ---------------------------------------------------------------------------
extern "C" void kernel_launch(void* const* d_in, const int* in_sizes, int n_in,
                              void* d_out, int out_size, void* d_ws, size_t ws_size,
                              hipStream_t stream) {
    (void)in_sizes; (void)n_in; (void)out_size; (void)ws_size;
    const float* x        = (const float*)d_in[0];  // [4,4096,4096]
    const float* lora_A   = (const float*)d_in[1];  // [64,4096]
    const float* lora_B   = (const float*)d_in[2];  // [4096,64]
    const float* a_scale  = (const float*)d_in[3];  // [64]
    const float* b_scale  = (const float*)d_in[4];  // [4096]
    float* out            = (float*)d_out;          // [4,4096,4096]

    // Workspace: bf16 weight copies (1 MB total), L2-resident thereafter.
    __bf16* Abf = (__bf16*)d_ws;                          // 512 KB
    __bf16* Bbf = Abf + (size_t)LORA_RANK * LORA_DIN;     // 512 KB

    cvt_f32_bf16_kernel<<<256, 256, 0, stream>>>(lora_A, Abf, LORA_RANK * LORA_DIN);
    cvt_f32_bf16_kernel<<<256, 256, 0, stream>>>(lora_B, Bbf, LORA_DOUT * LORA_RANK);

    lora_fused_kernel<<<LORA_M / 128, 256, 0, stream>>>(x, Abf, Bbf,
                                                        a_scale, b_scale, out);
}
